// RelLearnableMultiHeadAttn_42047729828000
// MI455X (gfx1250) — matrix operand order repaired and checked by reading, not compiled, not run on hardware
//
#include <hip/hip_runtime.h>

// Transformer-XL relative multi-head attention block for gfx1250 (MI455X).
// Pipeline: [w @ W_qkv] -> flash attention w/ analytic rel-shift -> [@ W_o]
// -> residual + LayerNorm.  All matmuls use v_wmma_f32_16x16x32_bf16
// (f32 inputs converted to bf16, f32 accumulate).

#define QLEN   2048
#define BSZ    2
#define D_MODEL 1024
#define N_HEAD 16
#define D_HEAD 64
#define SCALE  0.125f        // 1/sqrt(64)
#define LN_EPS 1e-5f

typedef __attribute__((ext_vector_type(16))) __bf16 v16bf;
typedef __attribute__((ext_vector_type(8)))  float  v8f;

union BFrag { unsigned u[8]; v16bf v; };

__device__ __forceinline__ unsigned pk(float a, float b) {
  // round-to-nearest-even f32 -> bf16, pack two into a dword (low = first)
  unsigned ua = __float_as_uint(a); ua = (ua + 0x7FFFu + ((ua >> 16) & 1u)) >> 16;
  unsigned ub = __float_as_uint(b); ub = (ub + 0x7FFFu + ((ub >> 16) & 1u)) >> 16;
  return (ua & 0xFFFFu) | (ub << 16);
}

__device__ __forceinline__ v8f zero8() {
  v8f z; 
#pragma unroll
  for (int i = 0; i < 8; ++i) z[i] = 0.0f;
  return z;
}

__device__ __forceinline__ v8f wmma_bf16(const BFrag& a, const BFrag& b, v8f c) {
  return __builtin_amdgcn_wmma_f32_16x16x32_bf16(false, a.v, false, b.v,
                                                 (short)0, c, false, false);
}

// ---------------------------------------------------------------------------
// Generic f32 GEMM, bf16 WMMA inside. C[M,N] = A[M,K] @ B[K,N], row-major.
// Block tile 128x128, 8 waves as 2(M) x 4(N), wave tile 64x32.
// M,N,K all multiples of 128/128/32 for our calls.
// ---------------------------------------------------------------------------
__global__ __launch_bounds__(256) void gemm_wmma(const float* __restrict__ A,
                                                 const float* __restrict__ B,
                                                 float* __restrict__ C,
                                                 int M, int N, int K) {
  __shared__ unsigned lA[128 * 17];   // [row m][k-pair], pair = (2c,2c+1)
  __shared__ unsigned lB[128 * 17];   // [col n][k-pair], pair = (2kp,2kp+1)
  const int tid  = threadIdx.x;
  const int lane = tid & 31, h = lane >> 4, ln16 = lane & 15;
  const int wid  = tid >> 5;
  const int bm = blockIdx.y * 128, bn = blockIdx.x * 128;
  const int wm0 = (wid >> 2) * 64, wn0 = (wid & 3) * 32;

  v8f acc[4][2];
#pragma unroll
  for (int sm = 0; sm < 4; ++sm)
#pragma unroll
    for (int sn = 0; sn < 2; ++sn) acc[sm][sn] = zero8();

  for (int k0 = 0; k0 < K; k0 += 32) {
    // stage A block [128 x 32] as bf16 pairs
#pragma unroll
    for (int idx = tid; idx < 2048; idx += 256) {
      int r = idx >> 4, c2 = idx & 15;
      const float* p = A + (size_t)(bm + r) * K + k0 + 2 * c2;
      lA[r * 17 + c2] = pk(p[0], p[1]);
    }
    // stage B block [32 x 128] as bf16 pairs along k
#pragma unroll
    for (int idx = tid; idx < 2048; idx += 256) {
      int kp = idx >> 7, n = idx & 127;
      const float* p = B + (size_t)(k0 + 2 * kp) * N + bn + n;
      lB[n * 17 + kp] = pk(p[0], p[N]);
    }
    __syncthreads();

    BFrag fA[4], fB[2];
#pragma unroll
    for (int sm = 0; sm < 4; ++sm) {
      int m = wm0 + sm * 16 + ln16;
#pragma unroll
      for (int p = 0; p < 4; ++p) fA[sm].u[p]     = lA[m * 17 + 8 * h + p];
#pragma unroll
      for (int p = 0; p < 4; ++p) fA[sm].u[4 + p] = lA[m * 17 + 8 * h + 4 + p];
    }
#pragma unroll
    for (int sn = 0; sn < 2; ++sn) {
      int n = wn0 + sn * 16 + ln16;
#pragma unroll
      for (int p = 0; p < 8; ++p) fB[sn].u[p] = lB[n * 17 + 8 * h + p];
    }
#pragma unroll
    for (int sm = 0; sm < 4; ++sm)
#pragma unroll
      for (int sn = 0; sn < 2; ++sn)
        acc[sm][sn] = wmma_bf16(fA[sm], fB[sn], acc[sm][sn]);
    __syncthreads();
  }

#pragma unroll
  for (int sm = 0; sm < 4; ++sm)
#pragma unroll
    for (int sn = 0; sn < 2; ++sn) {
      int col = bn + wn0 + sn * 16 + ln16;
#pragma unroll
      for (int r = 0; r < 8; ++r) {
        int row = bm + wm0 + sm * 16 + r + 8 * h;
        C[(size_t)row * N + col] = acc[sm][sn][r];
      }
    }
}

// ---------------------------------------------------------------------------
// Flash attention with Transformer-XL relative shift.
// Grid: (QLEN/128, N_HEAD*BSZ); block 256 = 8 waves, each wave owns a 16-row
// query tile.  Keys processed in 32-wide chunks; V staged transposed in LDS
// (shared by all 8 waves).  For j<=i:
//   score = ((q_i+rwb)·k_j + q_i·r_emb[L-1-i+j] + r_bias[L-1-i+j]) * SCALE
// Masked (j>i) entries are -1e9.
// ---------------------------------------------------------------------------
__global__ __launch_bounds__(256) void attn_flash(
    const float* __restrict__ WH,     // w_heads [4096 x 3072]
    const float* __restrict__ REMB,   // [2048,16,64]
    const float* __restrict__ RWB,    // [16,64]
    const float* __restrict__ RBIAS,  // [2048,16]
    float* __restrict__ AV) {         // attn_vec [4096 x 1024]
  __shared__ unsigned lVT[64 * 17];   // V transposed: [d][j-pair] bf16
  __shared__ float    lS[8][16 * 33]; // AC chunk per wave
  __shared__ float    lT[8][16 * 49]; // band product per wave
  __shared__ unsigned lP[8][16 * 17]; // probs packed bf16 per wave
  __shared__ float    lAl[8][16];
  __shared__ float    lL[8][16];

  const int tid  = threadIdx.x;
  const int lane = tid & 31, h = lane >> 4, ln16 = lane & 15;
  const int wid  = tid >> 5;
  const int head = blockIdx.y >> 1, b = blockIdx.y & 1;
  const int wg_i0 = blockIdx.x * 128;
  const int it0   = wg_i0 + wid * 16;

  // Q fragments (2 k-steps of 32 over d_head=64), plain and +r_w_bias.
  BFrag fQ[2], fQb[2];
  {
    const int i = it0 + ln16;
    const float* qrow = WH + (size_t)(i * BSZ + b) * 3072 + head * 64;
    const float* wb   = RWB + head * 64;
#pragma unroll
    for (int kk = 0; kk < 2; ++kk)
#pragma unroll
      for (int p = 0; p < 8; ++p) {
        int d = kk * 32 + 16 * h + 2 * p;
        float a0 = qrow[d], a1 = qrow[d + 1];
        fQ[kk].u[p]  = pk(a0, a1);
        fQb[kk].u[p] = pk(a0 + wb[d], a1 + wb[d + 1]);
      }
  }

  v8f accO[4];
#pragma unroll
  for (int tv = 0; tv < 4; ++tv) accO[tv] = zero8();
  float mrun = -3.0e38f, lrun = 0.0f;   // meaningful in lanes 0..15

  const int jmax = wg_i0 + 128;         // uniform across workgroup
  for (int j0 = 0; j0 < jmax; j0 += 32) {
    // ---- stage V chunk [32 x 64] transposed as bf16 j-pairs ----
#pragma unroll
    for (int idx = tid; idx < 1024; idx += 256) {
      int jp = idx >> 6, d = idx & 63;
      const float* vp = WH + (size_t)((j0 + 2 * jp) * BSZ + b) * 3072
                        + 2048 + head * 64 + d;
      lVT[d * 17 + jp] = pk(vp[0], vp[3072 * BSZ]);
    }
    __syncthreads();

    // ---- AC = (Q+rwb) @ K^T for 2 key sub-tiles ----
#pragma unroll
    for (int jt = 0; jt < 2; ++jt) {
      v8f s = zero8();
      const int j = j0 + jt * 16 + ln16;
#pragma unroll
      for (int kk = 0; kk < 2; ++kk) {
        BFrag fK;
        const float* kr = WH + (size_t)(j * BSZ + b) * 3072 + 1024
                          + head * 64 + kk * 32 + 16 * h;
#pragma unroll
        for (int p = 0; p < 8; ++p) fK.u[p] = pk(kr[2 * p], kr[2 * p + 1]);
        s = wmma_bf16(fQb[kk], fK, s);
      }
#pragma unroll
      for (int r = 0; r < 8; ++r)
        lS[wid][(r + 8 * h) * 33 + jt * 16 + ln16] = s[r];
    }

    // ---- band product T = Q @ r_emb[rb..rb+47]^T (rel-shift source) ----
    const int rb = QLEN - 16 - it0 + j0;
#pragma unroll
    for (int tt = 0; tt < 3; ++tt) {
      v8f s = zero8();
      const int rr = rb + tt * 16 + ln16;
      const bool ok = (rr >= 0) && (rr < QLEN);
#pragma unroll
      for (int kk = 0; kk < 2; ++kk) {
        BFrag fR;
        const float* rrow = REMB + ((size_t)(ok ? rr : 0) * 16 + head) * 64
                            + kk * 32 + 16 * h;
#pragma unroll
        for (int p = 0; p < 8; ++p)
          fR.u[p] = ok ? pk(rrow[2 * p], rrow[2 * p + 1]) : 0u;
        s = wmma_bf16(fQ[kk], fR, s);
      }
#pragma unroll
      for (int r = 0; r < 8; ++r)
        lT[wid][(r + 8 * h) * 49 + tt * 16 + ln16] = s[r];
    }
    __syncthreads();

    // ---- online softmax row pass: lane r handles query row it0+r ----
    if (lane < 16) {
      const int i = it0 + lane;
      float sc[32];
      float cm = -3.0e38f;
#pragma unroll
      for (int dj = 0; dj < 32; ++dj) {
        int j = j0 + dj;
        float v;
        if (j <= i) {
          int c = dj + 15 - lane;                        // band column
          float t   = lT[wid][lane * 49 + c];
          float rbv = RBIAS[(size_t)(rb + c) * 16 + head];
          v = (lS[wid][lane * 33 + dj] + t + rbv) * SCALE;
        } else {
          v = -1.0e9f;
        }
        sc[dj] = v;
        cm = fmaxf(cm, v);
      }
      float nm = fmaxf(mrun, cm);
      float alpha = __expf(mrun - nm);
      float csum = 0.0f;
#pragma unroll
      for (int t2 = 0; t2 < 16; ++t2) {
        float p0 = __expf(sc[2 * t2]     - nm);
        float p1 = __expf(sc[2 * t2 + 1] - nm);
        csum += p0 + p1;
        lP[wid][lane * 17 + t2] = pk(p0, p1);
      }
      lrun = lrun * alpha + csum;
      mrun = nm;
      lAl[wid][lane] = alpha;
    }
    __syncthreads();

    // ---- rescale O, then O += P @ V via WMMA ----
    float al[8];
#pragma unroll
    for (int r = 0; r < 8; ++r) al[r] = lAl[wid][r + 8 * h];
    BFrag fP;
#pragma unroll
    for (int p = 0; p < 4; ++p) fP.u[p]     = lP[wid][ln16 * 17 + 8 * h + p];
#pragma unroll
    for (int p = 0; p < 4; ++p) fP.u[4 + p] = lP[wid][ln16 * 17 + 8 * h + 4 + p];
#pragma unroll
    for (int tv = 0; tv < 4; ++tv) {
      BFrag fV;
#pragma unroll
      for (int p = 0; p < 8; ++p)
        fV.u[p] = lVT[(tv * 16 + ln16) * 17 + 8 * h + p];
      v8f o = accO[tv];
#pragma unroll
      for (int r = 0; r < 8; ++r) o[r] *= al[r];
      accO[tv] = wmma_bf16(fP, fV, o);
    }
    __syncthreads();
  }

  if (lane < 16) lL[wid][lane] = lrun;
  __syncthreads();
  float li[8];
#pragma unroll
  for (int r = 0; r < 8; ++r) li[r] = 1.0f / lL[wid][r + 8 * h];
#pragma unroll
  for (int tv = 0; tv < 4; ++tv) {
    int col = head * 64 + tv * 16 + ln16;
#pragma unroll
    for (int r = 0; r < 8; ++r) {
      int i = it0 + r + 8 * h;
      AV[(size_t)(i * BSZ + b) * 1024 + col] = accO[tv][r] * li[r];
    }
  }
}

// ---------------------------------------------------------------------------
// y = w + attn_out; out = LayerNorm(y) * g + b.  One block per row of 1024.
// ---------------------------------------------------------------------------
__global__ __launch_bounds__(256) void resid_ln(const float* __restrict__ W,
                                                const float* __restrict__ AO,
                                                const float* __restrict__ G,
                                                const float* __restrict__ Bb,
                                                float* __restrict__ OUT) {
  __shared__ float red[256];
  const int row = blockIdx.x;
  const float* wr = W  + (size_t)row * D_MODEL;
  const float* ar = AO + (size_t)row * D_MODEL;
  float* orow = OUT + (size_t)row * D_MODEL;
  float y[4];
  float s = 0.0f;
#pragma unroll
  for (int t = 0; t < 4; ++t) {
    int c = threadIdx.x + 256 * t;
    y[t] = wr[c] + ar[c];
    s += y[t];
  }
  red[threadIdx.x] = s;
  __syncthreads();
  for (int o = 128; o > 0; o >>= 1) {
    if (threadIdx.x < o) red[threadIdx.x] += red[threadIdx.x + o];
    __syncthreads();
  }
  float mu = red[0] / (float)D_MODEL;
  __syncthreads();
  float vs = 0.0f;
#pragma unroll
  for (int t = 0; t < 4; ++t) { float d = y[t] - mu; vs += d * d; }
  red[threadIdx.x] = vs;
  __syncthreads();
  for (int o = 128; o > 0; o >>= 1) {
    if (threadIdx.x < o) red[threadIdx.x] += red[threadIdx.x + o];
    __syncthreads();
  }
  float rinv = rsqrtf(red[0] / (float)D_MODEL + LN_EPS);
#pragma unroll
  for (int t = 0; t < 4; ++t) {
    int c = threadIdx.x + 256 * t;
    orow[c] = (y[t] - mu) * rinv * G[c] + Bb[c];
  }
}

// ---------------------------------------------------------------------------
extern "C" void kernel_launch(void* const* d_in, const int* in_sizes, int n_in,
                              void* d_out, int out_size, void* d_ws, size_t ws_size,
                              hipStream_t stream) {
  (void)in_sizes; (void)n_in; (void)out_size; (void)ws_size;
  const float* w     = (const float*)d_in[0];
  const float* remb  = (const float*)d_in[1];
  const float* rwb   = (const float*)d_in[2];
  const float* rbias = (const float*)d_in[3];
  // d_in[4] = attn_mask: causal mask applied analytically in-kernel
  const float* wqkv  = (const float*)d_in[5];
  const float* wo    = (const float*)d_in[6];
  const float* lng   = (const float*)d_in[7];
  const float* lnb   = (const float*)d_in[8];

  float* wheads = (float*)d_ws;                       // 4096*3072 f32
  float* avec   = wheads + (size_t)4096 * 3072;       // 4096*1024 f32
  float* aout   = avec   + (size_t)4096 * 1024;       // 4096*1024 f32

  gemm_wmma<<<dim3(3072 / 128, 4096 / 128), 256, 0, stream>>>(
      w, wqkv, wheads, 4096, 3072, 1024);
  attn_flash<<<dim3(QLEN / 128, N_HEAD * BSZ), 256, 0, stream>>>(
      wheads, remb, rwb, rbias, avec);
  gemm_wmma<<<dim3(1024 / 128, 4096 / 128), 256, 0, stream>>>(
      avec, wo, aout, 4096, 1024, 1024);
  resid_ln<<<dim3(4096), 256, 0, stream>>>(w, aout, lng, lnb, (float*)d_out);
}
